// RWKV_x070_64415919505552
// MI455X (gfx1250) — matrix operand order repaired and checked by reading, not compiled, not run on hardware
//
#include <hip/hip_runtime.h>
#include <hip/hip_bf16.h>
#include <math.h>

// RWKV-7 forward for MI455X (gfx1250): bf16 WMMA GEMMs + register-resident scan.
#define TDIM 4096
#define CDIM 2048
#define HDIM 32
#define NDIM 64

typedef __attribute__((ext_vector_type(16))) __bf16 v16bf;
typedef __attribute__((ext_vector_type(8)))  __bf16 v8bf;
typedef __attribute__((ext_vector_type(2)))  __bf16 v2bf;
typedef __attribute__((ext_vector_type(8)))  float  v8f;

// Hardware-rate transcendentals (v_rcp_f32 / v_rsq_f32, ~1 ulp).
__device__ __forceinline__ float rcp_fast(float x)  { return __builtin_amdgcn_rcpf(x); }
__device__ __forceinline__ float rsq_fast(float x)  { return __builtin_amdgcn_rsqf(x); }
__device__ __forceinline__ float sigmoidf_(float x) {
    return rcp_fast(1.0f + __expf(-x));
}
__device__ __forceinline__ float tanhf_fast(float x) {
    float e = __expf(2.0f * x);            // branch-free tanh
    return 1.0f - 2.0f * rcp_fast(e + 1.0f);
}

// ---------------------------------------------------------------------------
// GEMM: C[M,N] = act( mixA(A)[M,K] @ B[K,N] ), row-major f32 in/out, bf16
// WMMA (v_wmma_f32_16x16x32_bf16) with f32 accumulation.
//
// Block: 256 threads (8 wave32), 256x64 output tile, K-step 32.
// Wave w owns two 16-row strips (2w, 2w+1) and all 64 cols
//   => 8 WMMAs per K-step from 2 A frags x 4 B frags (1.5 ds_load_b128/WMMA).
//
// LDS tiles staged in *fragment order* ([strip][lane][16 elems]) so each lane
// loads its v16bf fragment as two ds_load_b128.
//   A frag (16-bit 16x32): lane<16 holds K {0..7,16..23}, lane>=16 {8..15,24..31}
//   B frag (16-bit 32x16): lane<16 holds K {0..7,16..23} col=lane, lane>=16 {8..15,24..31}
// ---------------------------------------------------------------------------
template <bool MIX, int ACT>
__global__ __launch_bounds__(256)
void gemm_bf16_wmma(const float* __restrict__ A, const float* __restrict__ B,
                    float* __restrict__ C, int M, int N, int K,
                    const float* __restrict__ mix,
                    const float* __restrict__ xprev)
{
    __shared__ __align__(32) __bf16 AsF[16 * 32 * 16];  // 16 strips x 32 lanes x 16 (16KB)
    __shared__ __align__(32) __bf16 BsF[ 4 * 32 * 16];  //  4 tiles  x 32 lanes x 16 ( 4KB)

    const int tid  = threadIdx.x;
    const int wv   = tid >> 5;
    const int lane = tid & 31;
    const int lr   = lane & 15;
    const bool hi  = (lane >= 16);
    const int m0 = blockIdx.y * 256;
    const int n0 = blockIdx.x * 64;

    // Staging coordinates (constant per thread).
    const int ast = tid >> 4;          // A strip 0..15 (thread stages row = tid)
    const int alr = tid & 15;
    const int bp  = tid >> 4;          // B row pair p: rows {2p, 2p+1}
    const int bc0 = (tid & 15) * 4;    // B col base
    const int bLaneHi = ((2 * bp) & 8) ? 16 : 0;
    const int bI0     = ((2 * bp) & 7) + ((2 * bp >= 16) ? 8 : 0);

    v8f acc[2][4];
    #pragma unroll
    for (int u = 0; u < 2; ++u)
        #pragma unroll
        for (int t = 0; t < 4; ++t)
            acc[u][t] = (v8f){0.f,0.f,0.f,0.f,0.f,0.f,0.f,0.f};

    const int KT = K >> 5;
    for (int kt = 0; kt < KT; ++kt) {
        const int kbase = kt << 5;

        // ---- Stage A (256x32): one full 32-wide row per thread ----
        {
            const int gm = m0 + tid;
            const float* __restrict__ Arow = A + (size_t)gm * K + kbase;
            float va[32];
            const float4* a4p = (const float4*)Arow;
            #pragma unroll
            for (int q = 0; q < 8; ++q) {
                float4 a4 = a4p[q];
                va[q*4+0] = a4.x; va[q*4+1] = a4.y; va[q*4+2] = a4.z; va[q*4+3] = a4.w;
            }
            if (MIX) {
                const float* __restrict__ Prow = (gm == 0) ? (xprev + kbase) : (Arow - K);
                const float* __restrict__ Mrow = mix + kbase;
                const float4* p4p = (const float4*)Prow;
                const float4* m4p = (const float4*)Mrow;
                #pragma unroll
                for (int q = 0; q < 8; ++q) {
                    float4 p4 = p4p[q], m4 = m4p[q];
                    va[q*4+0] += (p4.x - va[q*4+0]) * m4.x;
                    va[q*4+1] += (p4.y - va[q*4+1]) * m4.y;
                    va[q*4+2] += (p4.z - va[q*4+2]) * m4.z;
                    va[q*4+3] += (p4.w - va[q*4+3]) * m4.w;
                }
            }
            // 4 regions of 8 consecutive K-cols -> 4 contiguous 16B stores.
            #pragma unroll
            for (int r = 0; r < 4; ++r) {
                const int laneD = alr + ((r & 1) ? 16 : 0);
                const int i0    = (r >= 2) ? 8 : 0;
                v8bf t;
                #pragma unroll
                for (int e = 0; e < 8; ++e) t[e] = (__bf16)va[r*8 + e];
                *(v8bf*)&AsF[ast*512 + laneD*16 + i0] = t;
            }
        }
        // ---- Stage B (32x64): row pair x 4 cols -> packed b32 stores ----
        {
            const float* __restrict__ Brow0 = B + (size_t)(kbase + 2*bp) * N + n0 + bc0;
            const float* __restrict__ Brow1 = Brow0 + N;
            float4 r0 = *(const float4*)Brow0;
            float4 r1 = *(const float4*)Brow1;
            float b0[4] = {r0.x, r0.y, r0.z, r0.w};
            float b1[4] = {r1.x, r1.y, r1.z, r1.w};
            #pragma unroll
            for (int u = 0; u < 4; ++u) {
                const int cc    = bc0 + u;
                const int tile  = cc >> 4;
                const int laneD = (cc & 15) + bLaneHi;
                v2bf t; t[0] = (__bf16)b0[u]; t[1] = (__bf16)b1[u];
                *(v2bf*)&BsF[tile*512 + laneD*16 + bI0] = t;     // 4B store
            }
        }
        __syncthreads();

        // ---- Fragments (contiguous v16bf per lane) + 8 WMMAs ----
        v16bf af[2];
        af[0] = *(const v16bf*)&AsF[(wv*2 + 0)*512 + lane*16];
        af[1] = *(const v16bf*)&AsF[(wv*2 + 1)*512 + lane*16];
        #pragma unroll
        for (int tt = 0; tt < 4; ++tt) {
            const v16bf bfrag = *(const v16bf*)&BsF[tt*512 + lane*16];
            acc[0][tt] = __builtin_amdgcn_wmma_f32_16x16x32_bf16(
                            false, af[0], false, bfrag, (short)0, acc[0][tt], false, false);
            acc[1][tt] = __builtin_amdgcn_wmma_f32_16x16x32_bf16(
                            false, af[1], false, bfrag, (short)0, acc[1][tt], false, false);
        }
        __syncthreads();
    }

    // Epilogue. C/D layout: VGPR d, lane<16 => row d, lane>=16 => row d+8.
    #pragma unroll
    for (int u = 0; u < 2; ++u) {
        #pragma unroll
        for (int tt = 0; tt < 4; ++tt) {
            const int col = n0 + tt*16 + lr;
            #pragma unroll
            for (int d = 0; d < 8; ++d) {
                const int row = m0 + (wv*2 + u)*16 + d + (hi ? 8 : 0);
                float v = acc[u][tt][d];
                if (ACT == 1)      v = tanhf_fast(v);
                else if (ACT == 2) v = sigmoidf_(v);
                C[(size_t)row * N + col] = v;
            }
        }
    }
}

// ---------------------------------------------------------------------------
// Sequential RWKV-7 scan. One block per head, 64 threads (2 wave32).
// Thread i owns state row i (64 f32 registers). Per-step vectors broadcast
// through LDS; reductions done redundantly per-thread with float4 LDS reads.
// Next step's 6 global stream loads are software-pipelined so their latency
// hides under the current step's state update / GroupNorm arithmetic.
// ---------------------------------------------------------------------------
__global__ __launch_bounds__(64)
void rwkv7_scan(const float* __restrict__ rmat, const float* __restrict__ wl,
                const float* __restrict__ kmat, const float* __restrict__ al,
                const float* __restrict__ gmat, const float* __restrict__ vmat,
                const float* __restrict__ state_in,
                const float* __restrict__ w0, const float* __restrict__ a0,
                const float* __restrict__ k_k, const float* __restrict__ k_a,
                const float* __restrict__ r_k, const float* __restrict__ ln_w,
                const float* __restrict__ ln_b,
                float* __restrict__ ybuf, float* __restrict__ state_out)
{
    const int h   = blockIdx.x;
    const int tid = threadIdx.x;           // 0..63 == state row i and channel j
    const int hc  = h * NDIM;

    float s[NDIM];
    #pragma unroll
    for (int j = 0; j < NDIM; ++j)
        s[j] = state_in[(size_t)h * NDIM * NDIM + tid * NDIM + j];

    __shared__ __align__(16) float rs[NDIM], ws[NDIM], ks[NDIM], kks[NDIM],
                                   as_[NDIM], ys[NDIM], ksr[NDIM], red[NDIM];

    const float w0v = w0[hc + tid], a0v = a0[hc + tid];
    const float kkc = k_k[hc + tid], kac = k_a[hc + tid];
    const float rkc = r_k[hc + tid];
    const float lnw = ln_w[hc + tid], lnb = ln_b[hc + tid];

    // Preload step 0 streams.
    size_t pbase = (size_t)hc + tid;
    float pr = rmat[pbase], pw = wl[pbase], pk = kmat[pbase],
          pa = al[pbase],   pv = vmat[pbase], pg = gmat[pbase];

    for (int t = 0; t < TDIM; ++t) {
        const size_t bcur = (size_t)t * CDIM + hc + tid;
        const float rv   = pr;
        const float wvv  = __expf(-0.6065306597f * sigmoidf_(w0v + pw));
        const float kraw = pk;
        const float av   = sigmoidf_(a0v + pa);
        const float vv   = pv;
        const float gv   = pg;
        const float kkraw = kraw * kkc;
        const float kfin  = kraw * (1.0f + (av - 1.0f) * kac);

        __syncthreads();   // previous step's LDS consumers are done
        rs[tid] = rv;  ws[tid] = wvv;  ks[tid] = kfin;
        kks[tid] = kkraw;  as_[tid] = av;  red[tid] = kkraw * kkraw;
        ksr[tid] = kfin * rkc;
        __syncthreads();

        // Issue next step's loads early; latency hides under compute below.
        if (t + 1 < TDIM) {
            pbase += CDIM;
            pr = rmat[pbase]; pw = wl[pbase]; pk = kmat[pbase];
            pa = al[pbase];   pv = vmat[pbase]; pg = gmat[pbase];
        }

        float nrm = 0.f;
        #pragma unroll
        for (int q = 0; q < NDIM/4; ++q) {
            float4 r4 = *(const float4*)&red[q*4];
            nrm += (r4.x + r4.y) + (r4.z + r4.w);
        }
        const float inv = rcp_fast(fmaxf(__builtin_amdgcn_sqrtf(nrm), 1e-12f));

        float sa = 0.f;
        #pragma unroll
        for (int q = 0; q < NDIM/4; ++q) {
            float4 k4 = *(const float4*)&kks[q*4];
            sa += s[q*4+0]*k4.x + s[q*4+1]*k4.y + s[q*4+2]*k4.z + s[q*4+3]*k4.w;
        }
        sa *= inv;

        float y = 0.f;
        #pragma unroll
        for (int q = 0; q < NDIM/4; ++q) {
            float4 k4 = *(const float4*)&kks[q*4];
            float4 w4 = *(const float4*)&ws[q*4];
            float4 a4 = *(const float4*)&as_[q*4];
            float4 kf = *(const float4*)&ks[q*4];
            float4 r4 = *(const float4*)&rs[q*4];
            float s0 = s[q*4+0]*w4.x - sa*(k4.x*inv*a4.x) + vv*kf.x;
            float s1 = s[q*4+1]*w4.y - sa*(k4.y*inv*a4.y) + vv*kf.y;
            float s2 = s[q*4+2]*w4.z - sa*(k4.z*inv*a4.z) + vv*kf.z;
            float s3 = s[q*4+3]*w4.w - sa*(k4.w*inv*a4.w) + vv*kf.w;
            s[q*4+0] = s0; s[q*4+1] = s1; s[q*4+2] = s2; s[q*4+3] = s3;
            y += s0*r4.x + s1*r4.y + s2*r4.z + s3*r4.w;
        }
        ys[tid] = y;
        __syncthreads();

        // GroupNorm over the head + bonus dot (redundant per-thread reductions).
        float mu = 0.f;
        #pragma unroll
        for (int q = 0; q < NDIM/4; ++q) {
            float4 y4 = *(const float4*)&ys[q*4];
            mu += (y4.x + y4.y) + (y4.z + y4.w);
        }
        mu *= (1.0f / NDIM);
        float var = 0.f;
        #pragma unroll
        for (int q = 0; q < NDIM/4; ++q) {
            float4 y4 = *(const float4*)&ys[q*4];
            float d0 = y4.x-mu, d1 = y4.y-mu, d2 = y4.z-mu, d3 = y4.w-mu;
            var += d0*d0 + d1*d1 + d2*d2 + d3*d3;
        }
        var *= (1.0f / NDIM);
        const float yn = (y - mu) * rsq_fast(var + 0.00064f) * lnw + lnb;

        float dot = 0.f;
        #pragma unroll
        for (int q = 0; q < NDIM/4; ++q) {
            float4 r4 = *(const float4*)&rs[q*4];
            float4 kr = *(const float4*)&ksr[q*4];
            dot += r4.x*kr.x + r4.y*kr.y + r4.z*kr.z + r4.w*kr.w;
        }
        const float bonus = dot * vv;

        ybuf[bcur] = (yn + bonus) * gv;
    }

    #pragma unroll
    for (int j = 0; j < NDIM; ++j)
        state_out[(size_t)h * NDIM * NDIM + tid * NDIM + j] = s[j];
}

__global__ void copy_last_row(const float* __restrict__ x, float* __restrict__ dst)
{
    int i = blockIdx.x * 256 + threadIdx.x;
    if (i < CDIM) dst[i] = x[(size_t)(TDIM - 1) * CDIM + i];
}

// ---------------------------------------------------------------------------
static inline void launch_gemm(dim3 grid, hipStream_t s,
                               const float* A, const float* B, float* C,
                               int M, int N, int K,
                               const float* mix, const float* xprev, int act)
{
    dim3 blk(256);
    if (mix) {
        if (act == 1)
            gemm_bf16_wmma<true, 1><<<grid, blk, 0, s>>>(A, B, C, M, N, K, mix, xprev);
        else if (act == 2)
            gemm_bf16_wmma<true, 2><<<grid, blk, 0, s>>>(A, B, C, M, N, K, mix, xprev);
        else
            gemm_bf16_wmma<true, 0><<<grid, blk, 0, s>>>(A, B, C, M, N, K, mix, xprev);
    } else {
        gemm_bf16_wmma<false, 0><<<grid, blk, 0, s>>>(A, B, C, M, N, K, nullptr, nullptr);
    }
}

extern "C" void kernel_launch(void* const* d_in, const int* in_sizes, int n_in,
                              void* d_out, int out_size, void* d_ws, size_t ws_size,
                              hipStream_t stream)
{
    (void)in_sizes; (void)n_in; (void)out_size; (void)ws_size;
    const float* x     = (const float*)d_in[0];
    const float* xprev = (const float*)d_in[1];
    const float* state = (const float*)d_in[2];
    const float* x_r   = (const float*)d_in[3];
    const float* x_w   = (const float*)d_in[4];
    const float* x_k   = (const float*)d_in[5];
    const float* x_v   = (const float*)d_in[6];
    const float* x_a   = (const float*)d_in[7];
    const float* x_g   = (const float*)d_in[8];
    const float* w0    = (const float*)d_in[9];
    const float* w1    = (const float*)d_in[10];
    const float* w2    = (const float*)d_in[11];
    const float* a0    = (const float*)d_in[12];
    const float* a1    = (const float*)d_in[13];
    const float* a2    = (const float*)d_in[14];
    const float* g1    = (const float*)d_in[15];
    const float* g2    = (const float*)d_in[16];
    const float* k_k   = (const float*)d_in[17];
    const float* k_a   = (const float*)d_in[18];
    const float* r_k   = (const float*)d_in[19];
    const float* R_    = (const float*)d_in[20];
    const float* K_    = (const float*)d_in[21];
    const float* V_    = (const float*)d_in[22];
    const float* O_    = (const float*)d_in[23];
    const float* ln_w  = (const float*)d_in[24];
    const float* ln_b  = (const float*)d_in[25];

    const size_t TC = (size_t)TDIM * CDIM;
    // Output tuple, flat in return order: out [T,C], x_last [C],
    // state [H,N,N], v_first [T,C].
    float* out        = (float*)d_out;
    float* out_y      = out;
    float* out_xlast  = out + TC;
    float* out_state  = out_xlast + CDIM;
    float* out_vfirst = out_state + (size_t)HDIM * NDIM * NDIM;

    float* ws   = (float*)d_ws;
    float* wsr  = ws;                              // r = xr @ R_
    float* wsk  = wsr  + TC;                       // k = xk @ K_
    float* wswl = wsk  + TC;                       // w LoRA output
    float* wsal = wswl + TC;                       // a LoRA output
    float* wsg  = wsal + TC;                       // g gate
    float* wsy  = wsg  + TC;                       // (yn + bonus) * g
    float* wswt = wsy  + TC;                       // tanh(xw @ w1)   [T,64]
    float* wsat = wswt + (size_t)TDIM * 64;        // xa @ a1         [T,64]
    float* wsgt = wsat + (size_t)TDIM * 64;        // sigmoid(xg@g1)  [T,128]

    dim3 gBig(CDIM / 64, TDIM / 256);
    dim3 gL64(1, TDIM / 256);
    dim3 gL128(2, TDIM / 256);

    // Projections (token-shift mixing fused into A-tile loads).
    launch_gemm(gBig,  stream, x, R_, wsr,        TDIM, CDIM, CDIM, x_r, xprev, 0);
    launch_gemm(gBig,  stream, x, K_, wsk,        TDIM, CDIM, CDIM, x_k, xprev, 0);
    launch_gemm(gBig,  stream, x, V_, out_vfirst, TDIM, CDIM, CDIM, x_v, xprev, 0);
    // w LoRA: tanh(xw @ w1) @ w2
    launch_gemm(gL64,  stream, x, w1, wswt, TDIM, 64,   CDIM, x_w, xprev, 1);
    launch_gemm(gBig,  stream, wswt, w2, wswl, TDIM, CDIM, 64, nullptr, nullptr, 0);
    // a LoRA: (xa @ a1) @ a2   (sigmoid(a0 + .) applied in scan)
    launch_gemm(gL64,  stream, x, a1, wsat, TDIM, 64,   CDIM, x_a, xprev, 0);
    launch_gemm(gBig,  stream, wsat, a2, wsal, TDIM, CDIM, 64, nullptr, nullptr, 0);
    // g gate: sigmoid(xg @ g1) @ g2
    launch_gemm(gL128, stream, x, g1, wsgt, TDIM, 128,  CDIM, x_g, xprev, 2);
    launch_gemm(gBig,  stream, wsgt, g2, wsg, TDIM, CDIM, 128, nullptr, nullptr, 0);

    // Sequential state scan + GroupNorm + bonus + gate.
    rwkv7_scan<<<dim3(HDIM), dim3(64), 0, stream>>>(
        wsr, wswl, wsk, wsal, wsg, out_vfirst, state,
        w0, a0, k_k, k_a, r_k, ln_w, ln_b, wsy, out_state);

    // Output projection.
    launch_gemm(gBig, stream, wsy, O_, out_y, TDIM, CDIM, CDIM, nullptr, nullptr, 0);
    copy_last_row<<<dim3((CDIM + 255) / 256), dim3(256), 0, stream>>>(x, out_xlast);
}
